// VN_Ori_DGCNN_22067541967080
// MI455X (gfx1250) — matrix operand order repaired and checked by reading, not compiled, not run on hardware
//
#include <hip/hip_runtime.h>
#include <hip/hip_bf16.h>
#include <math.h>

// ============================================================================
// VN-DGCNN forward for MI455X (gfx1250, wave32, WMMA).
//
// Roofline: ~30 GFLOP total -> far below the WMMA ceiling; the workload is
// data-movement bound and the per-hop tensors fit the 192MB L2. Strategy:
//  * Every GEMM-shaped op (kNN Gram matrices, all VN channel-mixing linears)
//    goes through v_wmma_f32_16x16x32_bf16 with fp32 accumulate.
//  * GEMM is templated on the column count (61440/3072/1024) so B-panel loads
//    are immediate-offset; split-K main loop is clamp/mask-free, only the
//    ragged tail tile pays for bounds handling. B fragments are written to
//    LDS as packed bf16 pairs with ds_store_b128; consumers fetch fragments
//    with two ds_load_b128 per operand.
//  * Irregular work (wave32 top-20 argmax selection via shfl_xor, gathers,
//    vector norms/BN, cross products, K=3 contractions) stays VALU.
// Requires ws_size >= ~282 MB.
// ============================================================================

#define EPSV 1e-6f
#define NEGS 0.2f

typedef __attribute__((ext_vector_type(16))) __bf16 v16bf;
typedef __attribute__((ext_vector_type(8)))  float  v8f;

__device__ __forceinline__ unsigned bfround(float f) {
  union { float f; unsigned u; } cv; cv.f = f;
  return cv.u + 0x7FFFu + ((cv.u >> 16) & 1u);    // round-to-nearest-even
}
__device__ __forceinline__ __bf16 f2bf(float f) {
  unsigned short h = (unsigned short)(bfround(f) >> 16);
  __bf16 r;
  __builtin_memcpy(&r, &h, sizeof(r));
  return r;
}
__device__ __forceinline__ unsigned f2bf_pack(float lo, float hi) {
  return (bfround(lo) >> 16) | (bfround(hi) & 0xFFFF0000u);
}

// ----------------------------------------------------------------------------
// Generic batched GEMM:  C[z] = A[z] * B[z] (+ C[z] if beta)
//   A: M x K strides (asr, asc), batch stride asb  (covers W and Gram xf^T)
//   B: K x NC row-major, batch stride bsb.  NC compile-time, multiple of 128.
//   C: M x NC row-major, batch stride csb
// Block = 256 threads = 8 waves. Block tile 16(M) x 128(N); each wave owns one
// 16x16 C tile. K-step 32: unmasked main loop + one masked tail tile.
// LDS fragments in per-lane WMMA operand order (A and B both contiguous K):
//   A 16x32: lane = m + 16*g, element e -> K = 16g + e
//   B 32x16: lane = n + 16*g, element e -> K = 16g + e
//   C/D    : vgpr r, lane L -> M = r + 8*(L>>4), N = L&15
// ----------------------------------------------------------------------------
template <int NC>
__global__ void wmma_gemm_bf16(const float* __restrict__ A,
                               const float* __restrict__ Bm,
                               float* __restrict__ C,
                               int M, int K,
                               long asb, int asr, int asc,
                               long bsb, long csb, int beta)
{
  __shared__ __align__(16) unsigned As[256];        // 32 lanes * 8 dwords
  __shared__ __align__(16) unsigned Bs[8][256];     // per-wave B fragments

  const int tid  = threadIdx.x;
  const int lane = tid & 31;
  const int wave = tid >> 5;
  const int g    = lane >> 4;
  const int mr   = lane & 15;
  const int m0   = blockIdx.y << 4;
  const int n0   = blockIdx.x << 7;          // block column base (128 wide)
  const int nw   = n0 + (wave << 4);         // this wave's column base
  const long zb  = blockIdx.z;
  const float* Ab = A  + zb * asb;
  const float* Bb = Bm + zb * bsb;
  float*       Cb = C  + zb * csb;

  __bf16* AsH = (__bf16*)As;

  // A element mapping for this thread (2 elements), loop-invariant.
  int arI[2], akI[2];
  long abase[2];
  int aslot[2];
#pragma unroll
  for (int i = 0; i < 2; ++i) {
    int e = (tid << 1) + i;                  // 0..511
    if (asc == 1) { arI[i] = e >> 5; akI[i] = e & 31; }   // W: K-contiguous
    else          { arI[i] = e & 15; akI[i] = e >> 4; }   // Gram: M-contiguous
    int am  = m0 + arI[i];
    int ams = (am < M) ? am : (M - 1);       // clamped row (garbage rows are
    abase[i] = (long)ams * asr + (long)akI[i] * asc;  //  masked at C store)
    int ak = akI[i];
    int gg = (ak >> 4) & 1;
    aslot[i] = (((arI[i] + (gg << 4)) << 4) + (ak & 15));
  }
  // B mapping: fixed column, 16 consecutive K-rows -> one full fragment row.
  const int bc = tid & 127;                  // column 0..127 (coalesced)
  const int bg = tid >> 7;                   // row group: K rows [16*bg, 16*bg+16)
  const float* bp0 = Bb + (long)(bg << 4) * NC + (n0 + bc);
  unsigned* myB = &Bs[bc >> 4][((bc & 15) + (bg << 4)) << 3];

  v8f acc = {0.f, 0.f, 0.f, 0.f, 0.f, 0.f, 0.f, 0.f};
  if (beta) {
#pragma unroll
    for (int r = 0; r < 8; ++r) {
      int mm = m0 + r + (g << 3);
      if (mm < M) acc[r] = Cb[(long)mm * NC + nw + mr];
    }
  }

#define FRAG_WMMA()                                                       \
  do {                                                                    \
    v16bf fa, fb;                                                         \
    uint4 a0 = *(const uint4*)&As[lane << 3];                             \
    uint4 a1 = *(const uint4*)&As[(lane << 3) + 4];                       \
    __builtin_memcpy(&fa, &a0, 16);                                       \
    __builtin_memcpy((char*)&fa + 16, &a1, 16);                           \
    uint4 b0 = *(const uint4*)&Bs[wave][lane << 3];                       \
    uint4 b1 = *(const uint4*)&Bs[wave][(lane << 3) + 4];                 \
    __builtin_memcpy(&fb, &b0, 16);                                       \
    __builtin_memcpy((char*)&fb + 16, &b1, 16);                           \
    acc = __builtin_amdgcn_wmma_f32_16x16x32_bf16(false, fa, false, fb,   \
                                                  (short)0, acc, false, false); \
  } while (0)

  const int kmain = K & ~31;
  // ================= unmasked main loop (full 32-wide K tiles) =============
  for (int k0 = 0; k0 < kmain; k0 += 32) {
    float va0 = Ab[abase[0] + (long)k0 * asc];
    float va1 = Ab[abase[1] + (long)k0 * asc];
    const float* bp = bp0 + (long)k0 * NC;
    if (k0 + 32 < K) __builtin_prefetch(bp + 32L * NC, 0, 0);
    float vb[16];
#pragma unroll
    for (int i = 0; i < 16; ++i) vb[i] = bp[i * NC];   // imm-offset loads
    AsH[aslot[0]] = f2bf(va0);
    AsH[aslot[1]] = f2bf(va1);
    unsigned pk[8];
#pragma unroll
    for (int j = 0; j < 8; ++j) pk[j] = f2bf_pack(vb[2 * j], vb[2 * j + 1]);
    uint4 q0; q0.x = pk[0]; q0.y = pk[1]; q0.z = pk[2]; q0.w = pk[3];
    uint4 q1; q1.x = pk[4]; q1.y = pk[5]; q1.z = pk[6]; q1.w = pk[7];
    *(uint4*)myB       = q0;                           // ds_store_b128
    *(uint4*)(myB + 4) = q1;
    __syncthreads();
    FRAG_WMMA();
    __syncthreads();
  }
  // ================= masked tail tile (K % 32 != 0) ========================
  if (kmain < K) {
    const int k0 = kmain;
    float va[2];
#pragma unroll
    for (int i = 0; i < 2; ++i) {
      int ka  = k0 + akI[i];
      int kas = (ka < K) ? ka : (K - 1);
      va[i] = Ab[abase[i] + (long)(kas - akI[i]) * asc];
    }
    float vb[16];
#pragma unroll
    for (int i = 0; i < 16; ++i) {
      int ka  = k0 + (bg << 4) + i;
      int kas = (ka < K) ? ka : (K - 1);
      vb[i] = Bb[(long)kas * NC + (n0 + bc)];
    }
#pragma unroll
    for (int i = 0; i < 2; ++i) {
      int ka = k0 + akI[i];
      AsH[aslot[i]] = f2bf((ka < K) ? va[i] : 0.f);
    }
    unsigned pk[8];
#pragma unroll
    for (int j = 0; j < 8; ++j) {
      int k_lo = k0 + (bg << 4) + 2 * j;
      float lo = (k_lo     < K) ? vb[2 * j]     : 0.f;
      float hi = (k_lo + 1 < K) ? vb[2 * j + 1] : 0.f;
      pk[j] = f2bf_pack(lo, hi);
    }
    uint4 q0; q0.x = pk[0]; q0.y = pk[1]; q0.z = pk[2]; q0.w = pk[3];
    uint4 q1; q1.x = pk[4]; q1.y = pk[5]; q1.z = pk[6]; q1.w = pk[7];
    *(uint4*)myB       = q0;
    *(uint4*)(myB + 4) = q1;
    __syncthreads();
    FRAG_WMMA();
    __syncthreads();
  }
#undef FRAG_WMMA

#pragma unroll
  for (int r = 0; r < 8; ++r) {
    int mm = m0 + r + (g << 3);
    if (mm < M) Cb[(long)mm * NC + nw + mr] = acc[r];
  }
}

// ----------------------------------------------------------------------------
// sq[b][n] = sum_d xf[b][d][n]^2
// ----------------------------------------------------------------------------
__global__ void sqnorm_k(const float* __restrict__ xf, float* __restrict__ sq,
                         int D, int Np, int Bn)
{
  int t = blockIdx.x * blockDim.x + threadIdx.x;
  if (t >= Bn * Np) return;
  int b = t / Np, n = t - b * Np;
  const float* p = xf + (long)b * D * Np + n;
  float s = 0.f;
  for (int d = 0; d < D; ++d) { float v = p[(long)d * Np]; s += v * v; }
  sq[t] = s;
}

// ----------------------------------------------------------------------------
// top-k (k=20) of neg_dist row; one wave32 per (b,n); Np = 1024.
// neg_dist[n][m] = 2*inner[n][m] - sq[n] - sq[m], computed on the fly.
// 20 rounds of wave-wide argmax (shfl_xor), tie-break on lower index.
// ----------------------------------------------------------------------------
__global__ void knn_topk_k(const float* __restrict__ inner,
                           const float* __restrict__ sq,
                           int* __restrict__ idx, int kk)
{
  const int Np = 1024;
  int b = blockIdx.y, n = blockIdx.x, lane = threadIdx.x;
  const float* row = inner + ((long)b * Np + n) * Np;
  const float* sb  = sq + (long)b * Np;
  float sn = sb[n];
  float v[32];
#pragma unroll
  for (int i = 0; i < 32; ++i) {
    int m = lane + (i << 5);
    v[i] = 2.f * row[m] - sn - sb[m];
  }
  int* orow = idx + ((long)b * Np + n) * kk;
  for (int j = 0; j < kk; ++j) {
    float bv = -3.4e38f; int bi = 0x7fffffff;
#pragma unroll
    for (int i = 0; i < 32; ++i) {
      int m = lane + (i << 5);
      if (v[i] > bv) { bv = v[i]; bi = m; }
    }
    for (int s = 16; s > 0; s >>= 1) {
      float ov = __shfl_xor(bv, s, 32);
      int   oi = __shfl_xor(bi, s, 32);
      if (ov > bv || (ov == bv && oi < bi)) { bv = ov; bi = oi; }
    }
    if (lane == 0) orow[j] = bi;
    if ((bi & 31) == lane) v[bi >> 5] = -3.4e38f;   // knock out the winner
  }
}

// ----------------------------------------------------------------------------
// g[b, 0:C ,d,n,j] = x[b,c,d,idx] - x[b,c,d,n] ;  g[b, C:2C ,d,n,j] = x[b,c,d,n]
// ----------------------------------------------------------------------------
__global__ void gather_gf_k(const float* __restrict__ x, const int* __restrict__ idx,
                            float* __restrict__ g, int Bn, int C, int Np, int kk)
{
  long t = (long)blockIdx.x * blockDim.x + threadIdx.x;
  long tot = (long)Bn * C * Np * kk;
  if (t >= tot) return;
  int j = (int)(t % kk);
  long r = t / kk;
  int n = (int)(r % Np); r /= Np;
  int c = (int)(r % C);  int b = (int)(r / C);
  int m = idx[((long)b * Np + n) * kk + j];
#pragma unroll
  for (int d = 0; d < 3; ++d) {
    float ctr = x[(((long)b * C + c) * 3 + d) * Np + n];
    float ft  = x[(((long)b * C + c) * 3 + d) * Np + m];
    long o1 = ((((long)b * 2 * C + c)       * 3 + d) * Np + n) * kk + j;
    long o2 = ((((long)b * 2 * C + (C + c)) * 3 + d) * Np + n) * kk + j;
    g[o1] = ft - ctr;
    g[o2] = ctr;
  }
}

// ----------------------------------------------------------------------------
// Per-channel mean / variance of vector norms of x (B,C,3,S). One block per c.
// mv[c] = mean, mv[C+c] = population variance.
// ----------------------------------------------------------------------------
__global__ void bn_stats_k(const float* __restrict__ x, float* __restrict__ mv,
                           int Bn, int C, long S)
{
  __shared__ float r1[256], r2[256];
  int c = blockIdx.x;
  long tot = (long)Bn * S;
  float s1 = 0.f, s2 = 0.f;
  for (long t = threadIdx.x; t < tot; t += blockDim.x) {
    int b = (int)(t / S); long s = t - (long)b * S;
    const float* p = x + (((long)b * C + c) * 3) * S + s;
    float vx = p[0], vy = p[S], vz = p[2 * S];
    float nn = sqrtf(vx * vx + vy * vy + vz * vz) + EPSV;
    s1 += nn; s2 += nn * nn;
  }
  r1[threadIdx.x] = s1; r2[threadIdx.x] = s2;
  __syncthreads();
  for (int off = 128; off > 0; off >>= 1) {
    if ((int)threadIdx.x < off) {
      r1[threadIdx.x] += r1[threadIdx.x + off];
      r2[threadIdx.x] += r2[threadIdx.x + off];
    }
    __syncthreads();
  }
  if (threadIdx.x == 0) {
    float inv = 1.f / (float)tot;
    float mu = r1[0] * inv;
    mv[c]     = mu;
    mv[C + c] = r2[0] * inv - mu * mu;
  }
}

// ----------------------------------------------------------------------------
// out = VNLeakyReLU( (mv ? VNBatchNorm(p) : p), d ).
// p:(B,C,3,S); d:(B,Cd,3,S), Cd==1 means shared direction.
// y = p' - (1-NEG) * min(dot,0)/(dsq+eps) * d
// ----------------------------------------------------------------------------
__global__ void bn_leaky_k(const float* __restrict__ p, const float* __restrict__ dm,
                           const float* __restrict__ mv, float* __restrict__ out,
                           int Bn, int C, int Cd, long S)
{
  long t = (long)blockIdx.x * blockDim.x + threadIdx.x;
  long tot = (long)Bn * C * S;
  if (t >= tot) return;
  long s = t % S; long r = t / S;
  int c = (int)(r % C); int b = (int)(r / C);
  long i0 = (((long)b * C + c) * 3) * S + s;
  float px = p[i0], py = p[i0 + S], pz = p[i0 + 2 * S];
  if (mv) {
    float nn = sqrtf(px * px + py * py + pz * pz) + EPSV;
    float sc = (nn - mv[c]) * rsqrtf(mv[C + c] + 1e-5f) / nn;
    px *= sc; py *= sc; pz *= sc;
  }
  int dc = (Cd == 1) ? 0 : c;
  long j0 = (((long)b * Cd + dc) * 3) * S + s;
  float dx = dm[j0], dy = dm[j0 + S], dz = dm[j0 + 2 * S];
  float dot = px * dx + py * dy + pz * dz;
  float dsq = dx * dx + dy * dy + dz * dz;
  float f = (dot < 0.f) ? ((1.f - NEGS) * dot / (dsq + EPSV)) : 0.f;
  out[i0]         = px - f * dx;
  out[i0 + S]     = py - f * dy;
  out[i0 + 2 * S] = pz - f * dz;
}

// o[i] = mean_j h[i*kk + j]
__global__ void mean_last_k(const float* __restrict__ h, float* __restrict__ o,
                            long On, int kk)
{
  long t = (long)blockIdx.x * blockDim.x + threadIdx.x;
  if (t >= On) return;
  const float* p = h + t * kk;
  float s = 0.f;
  for (int j = 0; j < kk; ++j) s += p[j];
  o[t] = s / (float)kk;
}

// channel-block copy: dst(B,Cdst,S3) [co..co+Cs) <- src(B,Cs,S3)
__global__ void copy_ch_k(const float* __restrict__ src, float* __restrict__ dst,
                          int Bn, int Cs, int Cdst, int co, long S3)
{
  long t = (long)blockIdx.x * blockDim.x + threadIdx.x;
  long tot = (long)Bn * Cs * S3;
  if (t >= tot) return;
  long r = t % S3; long q = t / S3;
  int c = (int)(q % Cs); int b = (int)(q / Cs);
  dst[((long)b * Cdst + co + c) * S3 + r] = src[t];
}

// xim = normalize(j) x x   (cross product), layouts (B,C,3,Np)
__global__ void cross_jn_k(const float* __restrict__ x, const float* __restrict__ jv,
                           float* __restrict__ xim, int Bn, int C, int Np)
{
  long t = (long)blockIdx.x * blockDim.x + threadIdx.x;
  long tot = (long)Bn * C * Np;
  if (t >= tot) return;
  int n = (int)(t % Np); long r = t / Np;
  int c = (int)(r % C); int b = (int)(r / C);
  long i0 = (((long)b * C + c) * 3) * (long)Np + n;
  float jx = jv[i0], jy = jv[i0 + Np], jz = jv[i0 + 2 * Np];
  float nn = sqrtf(jx * jx + jy * jy + jz * jz) + EPSV;
  jx /= nn; jy /= nn; jz /= nn;
  float xx = x[i0], xy = x[i0 + Np], xz = x[i0 + 2 * Np];
  xim[i0]          = jy * xz - jz * xy;
  xim[i0 + Np]     = jz * xx - jx * xz;
  xim[i0 + 2 * Np] = jx * xy - jy * xx;
}

// feat[b,c,d] = meanh[b, c&511, d]  (concat(h, mean) then mean over N shortcut)
__global__ void feat_write_k(const float* __restrict__ mh, float* __restrict__ feat,
                             int Bn)
{
  int t = blockIdx.x * blockDim.x + threadIdx.x;
  int tot = Bn * 1024 * 3;
  if (t >= tot) return;
  int d = t % 3; int c = (t / 3) % 1024; int b = t / 3072;
  feat[t] = mh[((long)b * 512 + (c & 511)) * 3 + d];
}

// y[b,o,d] = sum_c W[o,c] * x[b,c,d]   (x,y layout (B,C,3))
__global__ void vn_lin_small_k(const float* __restrict__ W, const float* __restrict__ xin,
                               float* __restrict__ y, int Bn, int O, int C)
{
  int t = blockIdx.x * blockDim.x + threadIdx.x;
  int tot = Bn * O * 3;
  if (t >= tot) return;
  int d = t % 3; int o = (t / 3) % O; int b = t / (3 * O);
  const float* wr = W + (long)o * C;
  float s = 0.f;
  for (int c = 0; c < C; ++c) s += wr[c] * xin[((long)b * C + c) * 3 + d];
  y[t] = s;
}

// z0[b,d,o] = sum_c Wz[o,c] * z[b,c,d]
__global__ void z0_k(const float* __restrict__ Wz, const float* __restrict__ z,
                     float* __restrict__ z0, int Bn, int C)
{
  int t = blockIdx.x * blockDim.x + threadIdx.x;
  if (t >= Bn * 9) return;
  int o = t % 3; int d = (t / 3) % 3; int b = t / 9;
  float s = 0.f;
  for (int c = 0; c < C; ++c) s += Wz[o * C + c] * z[((long)b * C + c) * 3 + d];
  z0[((long)b * 3 + d) * 3 + o] = s;
}

// xs[b,i,k] = sum_j feat[b,i,j] * z0[b,j,k]
__global__ void xstd_k(const float* __restrict__ feat, const float* __restrict__ z0,
                       float* __restrict__ xs, int Bn)
{
  int t = blockIdx.x * blockDim.x + threadIdx.x;
  int tot = Bn * 1024 * 3;
  if (t >= tot) return;
  int kk = t % 3; int i = (t / 3) % 1024; int b = t / 3072;
  const float* fr = feat + ((long)b * 1024 + i) * 3;
  const float* zb = z0 + b * 9;
  xs[t] = fr[0] * zb[0 * 3 + kk] + fr[1] * zb[1 * 3 + kk] + fr[2] * zb[2 * 3 + kk];
}

// out[b,i,o] = xs[b,i,:] . W0[o,:] + b0[o]    (K=3, write-bound -> VALU)
__global__ void inv_k(const float* __restrict__ xs, const float* __restrict__ W0,
                      const float* __restrict__ b0, float* __restrict__ out, int Bn)
{
  long t = (long)blockIdx.x * blockDim.x + threadIdx.x;
  long tot = (long)Bn * 1024 * 1024;
  if (t >= tot) return;
  int o = (int)(t % 1024); long r = t / 1024;
  int i = (int)(r % 1024); int b = (int)(r / 1024);
  const float* xr = xs + ((long)b * 1024 + i) * 3;
  out[t] = xr[0] * W0[o * 3] + xr[1] * W0[o * 3 + 1] + xr[2] * W0[o * 3 + 2] + b0[o];
}

// ============================================================================
// Host orchestration
// ============================================================================
static inline int nblk(long n) { return (int)((n + 255) / 256); }

static inline void gemm(hipStream_t s, const float* A, const float* B, float* C,
                        int M, int Nc, int K, long asb, int asr, int asc,
                        long bsb, long csb, int beta, int batch)
{
  dim3 grid((Nc + 127) / 128, (M + 15) / 16, batch);
  if (Nc == 61440)
    wmma_gemm_bf16<61440><<<grid, 256, 0, s>>>(A, B, C, M, K, asb, asr, asc, bsb, csb, beta);
  else if (Nc == 3072)
    wmma_gemm_bf16<3072><<<grid, 256, 0, s>>>(A, B, C, M, K, asb, asr, asc, bsb, csb, beta);
  else
    wmma_gemm_bf16<1024><<<grid, 256, 0, s>>>(A, B, C, M, K, asb, asr, asc, bsb, csb, beta);
}

extern "C" void kernel_launch(void* const* d_in, const int* in_sizes, int n_in,
                              void* d_out, int out_size, void* d_ws, size_t ws_size,
                              hipStream_t stream)
{
  (void)in_sizes; (void)n_in; (void)out_size; (void)ws_size;
  const int B = 8, Np = 1024, KN = 20;
  const long Le = 3L * Np * KN;   // 61440 cols for edge-conv GEMMs
  const long Lp = 3L * Np;        // 3072 cols for per-point GEMMs

  const float* x    = (const float*)d_in[0];
  const float* Wf1  = (const float*)d_in[1];
  const float* Wd1  = (const float*)d_in[2];
  const float* Wf2  = (const float*)d_in[3];
  const float* Wd2  = (const float*)d_in[4];
  const float* Wf3  = (const float*)d_in[5];
  const float* Wd3  = (const float*)d_in[6];
  const float* Wf4  = (const float*)d_in[7];
  const float* Wd4  = (const float*)d_in[8];
  const float* Wf5  = (const float*)d_in[9];
  const float* Wd5  = (const float*)d_in[10];
  const float* Wl1  = (const float*)d_in[11];
  const float* Wl2  = (const float*)d_in[12];
  const float* Wr1  = (const float*)d_in[13];
  const float* Wi1  = (const float*)d_in[14];
  const float* Wc1  = (const float*)d_in[15];
  const float* Wl3  = (const float*)d_in[16];
  const float* Wl4  = (const float*)d_in[17];
  const float* Wr2  = (const float*)d_in[18];
  const float* Wi2  = (const float*)d_in[19];
  const float* Wc2  = (const float*)d_in[20];
  const float* Wf6  = (const float*)d_in[21];
  const float* Wd6  = (const float*)d_in[22];
  const float* Wv1f = (const float*)d_in[23];
  const float* Wv1d = (const float*)d_in[24];
  const float* Wv2f = (const float*)d_in[25];
  const float* Wv2d = (const float*)d_in[26];
  const float* Wz   = (const float*)d_in[27];
  const float* W0   = (const float*)d_in[28];
  const float* b0   = (const float*)d_in[29];

  float* out  = (float*)d_out;
  float* feat = out;                          // (B,1024,3)
  float* inv  = out + (long)B * 1024 * 3;     // (B,1024,1024)

  // ---- workspace bump allocation (floats) ----
  float* ws = (float*)d_ws;
  long off = 0;
  float* dist = ws + off; off += 8L * 1024 * 1024;   // Gram / later xb,jb
  float* sq   = ws + off; off += 8192;
  float* mv   = ws + off; off += 1024;               // 2 * 512 max channels
  int*   idx  = (int*)(ws + off); off += 163840;     // B*N*k
  float* g    = ws + off; off += 20643840;           // graph feat / comb,comb2
  float* t1   = ws + off; off += 12582912;           // p / xa,ja,xim / h6
  float* t2   = ws + off; off += 12582912;           // d / y / d6
  float* t3   = ws + off; off += 12582912;           // h / complex2 p,d
  float* x1   = ws + off; off += 516096;
  float* x2   = ws + off; off += 516096;
  float* x3   = ws + off; off += 516096;
  float* x123 = ws + off; off += 1548288;
  float* meanh= ws + off; off += 12288;
  float* z1   = ws + off; off += 12288;
  float* zd   = ws + off; off += 12288;
  float* z2   = ws + off; off += 6144;
  float* zz0  = ws + off; off += 72;
  float* xstd = ws + off; off += 24576;

  // ---- one EdgeConv block: kNN -> gather -> (VNlin+BN+leaky) x (1|2) -> mean_k
  auto edge_block = [&](const float* xin, int C, const float* Wfa, const float* Wda,
                        const float* Wfb, const float* Wdb, float* xout) {
    int D = C * 3;
    sqnorm_k<<<nblk((long)B * Np), 256, 0, stream>>>(xin, sq, D, Np, B);
    // Gram = xf^T xf  (WMMA; A = xf^T via strides asr=1, asc=Np)
    gemm(stream, xin, xin, dist, Np, Np, D,
         (long)D * Np, 1, Np, (long)D * Np, (long)Np * Np, 0, B);
    knn_topk_k<<<dim3(Np, B), 32, 0, stream>>>(dist, sq, idx, KN);
    gather_gf_k<<<nblk((long)B * C * Np * KN), 256, 0, stream>>>(xin, idx, g, B, C, Np, KN);
    int C2 = 2 * C;
    gemm(stream, Wfa, g, t1, 21, (int)Le, C2, 0, C2, 1, (long)C2 * Le, 21L * Le, 0, B);
    gemm(stream, Wda, g, t2, 21, (int)Le, C2, 0, C2, 1, (long)C2 * Le, 21L * Le, 0, B);
    bn_stats_k<<<21, 256, 0, stream>>>(t1, mv, B, 21, (long)Np * KN);
    bn_leaky_k<<<nblk((long)B * 21 * Np * KN), 256, 0, stream>>>(
        t1, t2, mv, t3, B, 21, 21, (long)Np * KN);
    const float* hsrc = t3;
    if (Wfb) {
      gemm(stream, Wfb, t3, t1, 21, (int)Le, 21, 0, 21, 1, 21L * Le, 21L * Le, 0, B);
      gemm(stream, Wdb, t3, t2, 21, (int)Le, 21, 0, 21, 1, 21L * Le, 21L * Le, 0, B);
      bn_stats_k<<<21, 256, 0, stream>>>(t1, mv, B, 21, (long)Np * KN);
      bn_leaky_k<<<nblk((long)B * 21 * Np * KN), 256, 0, stream>>>(
          t1, t2, mv, g, B, 21, 21, (long)Np * KN);   // g free -> reuse as h2
      hsrc = g;
    }
    mean_last_k<<<nblk((long)B * 21 * 3 * Np), 256, 0, stream>>>(
        hsrc, xout, (long)B * 21 * 3 * Np, KN);
  };

  edge_block(x,  1,  Wf1, Wd1, Wf2, Wd2, x1);
  edge_block(x1, 21, Wf3, Wd3, Wf4, Wd4, x2);
  edge_block(x2, 21, Wf5, Wd5, nullptr, nullptr, x3);

  // x123 = concat(x1,x2,x3) on channel axis
  copy_ch_k<<<nblk((long)B * 21 * Lp), 256, 0, stream>>>(x1, x123, B, 21, 63, 0,  Lp);
  copy_ch_k<<<nblk((long)B * 21 * Lp), 256, 0, stream>>>(x2, x123, B, 21, 63, 21, Lp);
  copy_ch_k<<<nblk((long)B * 21 * Lp), 256, 0, stream>>>(x3, x123, B, 21, 63, 42, Lp);

  const long szP = (long)B * 170 * Lp;   // 4,177,920 floats per 170-ch tensor
  float* xa   = t1;
  float* ja   = t1 + szP;
  float* ximb = t1 + 2 * szP;

  // complex block: p = Wr x + Wi (j_hat x x); d = Wc x; y = leaky(p,d) in-place p
  auto complex_block = [&](const float* xv, const float* jv, const float* Wr,
                           const float* Wi, const float* Wc, float* pb, float* db) {
    cross_jn_k<<<nblk((long)B * 170 * Np), 256, 0, stream>>>(xv, jv, ximb, B, 170, Np);
    gemm(stream, Wr, xv,   pb, 170, (int)Lp, 170, 0, 170, 1, 170L * Lp, 170L * Lp, 0, B);
    gemm(stream, Wi, ximb, pb, 170, (int)Lp, 170, 0, 170, 1, 170L * Lp, 170L * Lp, 1, B);
    gemm(stream, Wc, xv,   db, 170, (int)Lp, 170, 0, 170, 1, 170L * Lp, 170L * Lp, 0, B);
    bn_leaky_k<<<nblk((long)B * 170 * Np), 256, 0, stream>>>(
        pb, db, nullptr, pb, B, 170, 170, (long)Np);   // no BN in complex path
  };

  gemm(stream, Wl1, x123, xa, 170, (int)Lp, 63, 0, 63, 1, 63L * Lp, 170L * Lp, 0, B);
  gemm(stream, Wl2, x123, ja, 170, (int)Lp, 63, 0, 63, 1, 63L * Lp, 170L * Lp, 0, B);
  complex_block(xa, ja, Wr1, Wi1, Wc1, t2, t2 + szP);          // y -> t2

  float* comb = g;                                             // (B,340,3,N)
  copy_ch_k<<<nblk((long)B * 170 * Lp), 256, 0, stream>>>(xa, comb, B, 170, 340, 0,   Lp);
  copy_ch_k<<<nblk((long)B * 170 * Lp), 256, 0, stream>>>(t2, comb, B, 170, 340, 170, Lp);

  float* xb = dist;
  float* jb = dist + szP;
  gemm(stream, Wl3, comb, xb, 170, (int)Lp, 340, 0, 340, 1, 340L * Lp, 170L * Lp, 0, B);
  gemm(stream, Wl4, comb, jb, 170, (int)Lp, 340, 0, 340, 1, 340L * Lp, 170L * Lp, 0, B);
  complex_block(xb, jb, Wr2, Wi2, Wc2, t3, t3 + szP);          // y2 -> t3

  float* comb2 = g + 8355840;                                  // (B,340,3,N)
  copy_ch_k<<<nblk((long)B * 170 * Lp), 256, 0, stream>>>(xb, comb2, B, 170, 340, 0,   Lp);
  copy_ch_k<<<nblk((long)B * 170 * Lp), 256, 0, stream>>>(t3, comb2, B, 170, 340, 170, Lp);

  // conv6: 340 -> 512, BN + shared-direction leaky (Wd6 has 1 channel)
  float* h6 = t1;            // B*512*3*N = 12,582,912 floats, fits exactly
  float* d6 = t2;
  gemm(stream, Wf6, comb2, h6, 512, (int)Lp, 340, 0, 340, 1, 340L * Lp, 512L * Lp, 0, B);
  gemm(stream, Wd6, comb2, d6, 1,   (int)Lp, 340, 0, 340, 1, 340L * Lp, 1L * Lp,   0, B);
  bn_stats_k<<<512, 256, 0, stream>>>(h6, mv, B, 512, (long)Np);
  bn_leaky_k<<<nblk((long)B * 512 * Np), 256, 0, stream>>>(
      h6, d6, mv, h6, B, 512, 1, (long)Np);

  // pooling: feat = concat(mean_N h, mean_N h)
  mean_last_k<<<nblk((long)B * 512 * 3), 256, 0, stream>>>(h6, meanh, (long)B * 512 * 3, Np);
  feat_write_k<<<nblk((long)B * 1024 * 3), 256, 0, stream>>>(meanh, feat, B);

  // ---- VNStdFeature head (tiny; K=3 contractions stay VALU) ----
  vn_lin_small_k<<<nblk((long)B * 512 * 3), 256, 0, stream>>>(Wv1f, feat, z1, B, 512, 1024);
  vn_lin_small_k<<<nblk((long)B * 512 * 3), 256, 0, stream>>>(Wv1d, feat, zd, B, 512, 1024);
  bn_stats_k<<<512, 256, 0, stream>>>(z1, mv, B, 512, 1L);
  bn_leaky_k<<<nblk((long)B * 512), 256, 0, stream>>>(z1, zd, mv, z1, B, 512, 512, 1L);

  vn_lin_small_k<<<nblk((long)B * 256 * 3), 256, 0, stream>>>(Wv2f, z1, z2, B, 256, 512);
  vn_lin_small_k<<<nblk((long)B * 256 * 3), 256, 0, stream>>>(Wv2d, z1, zd, B, 256, 512);
  bn_stats_k<<<256, 256, 0, stream>>>(z2, mv, B, 256, 1L);
  bn_leaky_k<<<nblk((long)B * 256), 256, 0, stream>>>(z2, zd, mv, z2, B, 256, 256, 1L);

  z0_k<<<1, 128, 0, stream>>>(Wz, z2, zz0, B, 256);
  xstd_k<<<nblk((long)B * 1024 * 3), 256, 0, stream>>>(feat, zz0, xstd, B);
  inv_k<<<nblk((long)B * 1024 * 1024), 256, 0, stream>>>(xstd, W0, b0, inv, B);
}
